// MultiHeadedSelfAttention_36172214567578
// MI455X (gfx1250) — matrix operand order repaired and checked by reading, not compiled, not run on hardware
//
#include <hip/hip_runtime.h>

// ---------------------------------------------------------------------------
// Multi-head self-attention, MI455X (gfx1250, wave32, bf16 WMMA, f32 accum)
//   Phase 1: QKV projection  y = x @ W^T + b  -> Q (pre-scaled by log2(e)),
//            K (row major [S,32]) and V^T ([32,S]) in bf16 workspace.
//   Phase 2: flash attention, one wave per (b, h, 16-query tile).
//            Softmax runs on all 32 lanes directly in WMMA C-layout:
//            row reductions via __shfl_xor within 16-lane half-groups,
//            exponentials are raw exp2 (log2e folded into Q), stats (m, l,
//            alpha) live in registers in exactly the C-layout row order.
//            No __syncthreads(): the only LDS traffic is the per-wave
//            P transpose, which is wave-internal (DS ops are in-order).
// ---------------------------------------------------------------------------

typedef __attribute__((ext_vector_type(16))) __bf16 v16bf;
typedef __attribute__((ext_vector_type(8)))  __bf16 v8bf;
typedef __attribute__((ext_vector_type(8)))  float  v8f;

constexpr int Bsz = 4;
constexpr int Ssz = 4096;
constexpr int Dsz = 128;
constexpr int Hn  = 4;
constexpr int HD  = 32;                         // head dim
constexpr int NE  = Bsz * Hn * Ssz * HD;        // elems per Q/K/Vt tensor
constexpr float LOG2E = 1.44269504088896340736f;

__device__ inline v8f wmma_bf16(v16bf a, v16bf b, v8f c) {
    // D = A(16x32) x B(32x16) + C(16x16 f32)
    return __builtin_amdgcn_wmma_f32_16x16x32_bf16(
        false, a, false, b, (short)0, c, false, false);
}

// Fragment from two contiguous 8-element bf16 chunks (16B each).
__device__ inline v16bf load_frag2(const __bf16* p0, const __bf16* p1) {
    v8bf lo = *(const v8bf*)p0;
    v8bf hi = *(const v8bf*)p1;
    v16bf r;
#pragma unroll
    for (int i = 0; i < 8; ++i) { r[i] = lo[i]; r[i + 8] = hi[i]; }
    return r;
}
__device__ inline v16bf load_frag16(const __bf16* p) { return load_frag2(p, p + 8); }

// Convert two 8-float chunks to a bf16 fragment.
__device__ inline v16bf cvt_frag2(const float* p0, const float* p1) {
    v16bf r;
#pragma unroll
    for (int i = 0; i < 8; ++i) {
        r[i]     = (__bf16)p0[i];
        r[i + 8] = (__bf16)p1[i];
    }
    return r;
}
__device__ inline v16bf cvt_frag16(const float* p) { return cvt_frag2(p, p + 8); }

// ---------------------------------------------------------------------------
// Phase 1: QKV projection. One wave per (matrix m, 16-row tile rt).
//   3 * (B*S/16) = 3072 waves -> 384 blocks of 8 waves.
// A-layout (16-bit 16x32): lane L -> row L%16; halves 0..7 = k in
//   [8g..8g+7], halves 8..15 = k in [8g+16..8g+23], g = L>=16.
// B-layout (16-bit 32x16): lane L -> col L%16, same k split as A-layout.
// C/D f32: lane L -> col L%16, vgpr v -> row v + 8g.
// ---------------------------------------------------------------------------
__global__ __launch_bounds__(256) void qkv_proj_kernel(
    const float* __restrict__ x,
    const float* __restrict__ Wq, const float* __restrict__ bq,
    const float* __restrict__ Wk, const float* __restrict__ bk,
    const float* __restrict__ Wv, const float* __restrict__ bv,
    __bf16* __restrict__ Q, __bf16* __restrict__ K, __bf16* __restrict__ Vt)
{
    const int lane = threadIdx.x & 31;
    const int wid  = blockIdx.x * 8 + (threadIdx.x >> 5);
    const int m    = wid >> 10;          // 0=Q 1=K 2=V
    const int rt   = wid & 1023;         // 16-row tile in flattened [B*S]
    const int half = lane >> 4;
    const int l16  = lane & 15;

    const float* W    = (m == 0) ? Wq : (m == 1) ? Wk : Wv;
    const float* bias = (m == 0) ? bq : (m == 1) ? bk : bv;
    const float  oscl = (m == 0) ? LOG2E : 1.0f;   // fold log2(e) into Q

    // A fragments for the 4 K-steps of the 128-deep dot product.
    const float* xr = x + (size_t)(rt * 16 + l16) * Dsz;
    v16bf a[4];
#pragma unroll
    for (int ks = 0; ks < 4; ++ks) {
        const int base = ks * 32 + half * 8;
        a[ks] = cvt_frag2(xr + base, xr + base + 16);
    }

#pragma unroll
    for (int ct = 0; ct < 8; ++ct) {                 // 8 output col tiles of 16
        const int dcol = ct * 16;
        v8f acc = {};
#pragma unroll
        for (int ks = 0; ks < 4; ++ks) {
            // B[k][n] = W[n][k]: lane -> W row (dcol + l16), k in [8g..8g+7] and [8g+16..8g+23].
            const float* wr = W + (size_t)(dcol + l16) * Dsz + ks * 32 + half * 8;
            acc = wmma_bf16(a[ks], cvt_frag2(wr, wr + 16), acc);
        }
        const float bval = bias[dcol + l16];
        const int h   = ct >> 1;                     // head
        const int cih = (ct & 1) * 16 + l16;         // col within head
#pragma unroll
        for (int v = 0; v < 8; ++v) {
            const int fr = rt * 16 + v + half * 8;   // flat row
            const int b  = fr >> 12;                 // /S (S=4096)
            const int s  = fr & (Ssz - 1);
            const __bf16 val = (__bf16)((acc[v] + bval) * oscl);
            if (m == 0)
                Q[((size_t)(b * Hn + h) * Ssz + s) * HD + cih] = val;
            else if (m == 1)
                K[((size_t)(b * Hn + h) * Ssz + s) * HD + cih] = val;
            else
                Vt[((size_t)(b * Hn + h) * HD + cih) * Ssz + s] = val;
        }
    }
}

// ---------------------------------------------------------------------------
// Phase 2: flash attention. One wave per (b, h, 16-query tile):
//   B*H*(S/16) = 4096 waves -> 512 blocks of 8 waves, fully decoupled.
// Per iteration over 32 keys:
//   S0,S1 = Q.K^T (2 WMMAs, already in log2 domain)
//   row max / row sum via shfl_xor(1,2,4,8) inside 16-lane half-groups
//   P = exp2(S - m) in C-layout registers -> transposed LDS -> A-layout
//   O = P.V + alpha*O (2 WMMAs with C operand).
// ---------------------------------------------------------------------------
__global__ __launch_bounds__(256) void flash_attn_kernel(
    const __bf16* __restrict__ Q, const __bf16* __restrict__ K,
    const __bf16* __restrict__ Vt, float* __restrict__ out)
{
    __shared__ __bf16 pT[8][32][16];      // per-wave P, [key][query] (transposed)

    const int wv   = threadIdx.x >> 5;
    const int lane = threadIdx.x & 31;
    const int wid  = blockIdx.x * 8 + wv;
    const int bh   = wid >> 8;            // (b*H + h)
    const int qt   = wid & 255;           // query tile
    const int half = lane >> 4;           // g: accumulator row group
    const int l16  = lane & 15;

    const __bf16* Qp = Q  + (size_t)bh * Ssz * HD;
    const __bf16* Kp = K  + (size_t)bh * Ssz * HD;
    const __bf16* Vp = Vt + (size_t)bh * HD * Ssz;

    // Q fragment (held for the whole pass).
    const __bf16* qr = Qp + (size_t)(qt * 16 + l16) * HD + half * 8;
    const v16bf aq = load_frag2(qr, qr + 16);

    v8f o0 = {}, o1 = {};
    v8f mrun, lrun;                       // stats for rows v + 8*half
#pragma unroll
    for (int v = 0; v < 8; ++v) { mrun[v] = -__builtin_inff(); lrun[v] = 0.0f; }

    for (int kb = 0; kb < Ssz / 32; ++kb) {
        const int key0 = kb * 32;

        // ---- scores (log2 domain): two 16x16 tiles over 32 keys ---------
        const __bf16* k0 = Kp + (size_t)(key0 + l16) * HD + half * 8;
        const __bf16* k1 = k0 + 16 * HD;
        v8f z = {};
        const v8f s0 = wmma_bf16(aq, load_frag2(k0, k0 + 16), z);
        const v8f s1 = wmma_bf16(aq, load_frag2(k1, k1 + 16), z);

        if (kb + 1 < Ssz / 32)            // CDNA5 global_prefetch_b8
            __builtin_prefetch(Kp + (size_t)(key0 + 32 + l16) * HD, 0, 1);

        // ---- online softmax, all 32 lanes, C-layout native --------------
        v8f mnew, alpha;
#pragma unroll
        for (int v = 0; v < 8; ++v) {
            float t = fmaxf(s0[v], s1[v]);
            t = fmaxf(t, __shfl_xor(t, 1, 32));
            t = fmaxf(t, __shfl_xor(t, 2, 32));
            t = fmaxf(t, __shfl_xor(t, 4, 32));
            t = fmaxf(t, __shfl_xor(t, 8, 32));
            const float mn = fmaxf(mrun[v], t);
            mnew[v]  = mn;
            alpha[v] = exp2f(mrun[v] - mn);     // 0 on first tile (-inf)
        }
        v8bf pb0, pb1;
#pragma unroll
        for (int v = 0; v < 8; ++v) {
            const float p0 = exp2f(s0[v] - mnew[v]);
            const float p1 = exp2f(s1[v] - mnew[v]);
            pb0[v] = (__bf16)p0;
            pb1[v] = (__bf16)p1;
            float rs = p0 + p1;
            rs += __shfl_xor(rs, 1, 32);
            rs += __shfl_xor(rs, 2, 32);
            rs += __shfl_xor(rs, 4, 32);
            rs += __shfl_xor(rs, 8, 32);
            lrun[v] = lrun[v] * alpha[v] + rs;
            mrun[v] = mnew[v];
        }

        // ---- P: C-layout regs -> transposed LDS -> A-layout frag --------
        // Lane owns key column l16 (s0) / 16+l16 (s1), rows 8g..8g+7:
        // contiguous in [key][query] layout -> single b128 store each.
        *(v8bf*)&pT[wv][l16][half * 8]      = pb0;
        *(v8bf*)&pT[wv][16 + l16][half * 8] = pb1;
        __builtin_amdgcn_wave_barrier();    // keep store->load order (wave-local)
        v16bf pa;
#pragma unroll
        for (int i = 0; i < 8; ++i) {       // A-layout: lane -> query row l16
            pa[i]     = pT[wv][half * 8 + i][l16];
            pa[i + 8] = pT[wv][half * 8 + 16 + i][l16];
        }

        // ---- O accumulate: O = P.V + alpha*O ----------------------------
        v8f c0, c1;
#pragma unroll
        for (int v = 0; v < 8; ++v) { c0[v] = o0[v] * alpha[v]; c1[v] = o1[v] * alpha[v]; }

        // B operand columns come from V^T rows -> contiguous.
        const __bf16* v0p = Vp + (size_t)l16 * Ssz + key0 + half * 8;
        const __bf16* v1p = v0p + 16 * Ssz;
        o0 = wmma_bf16(pa, load_frag2(v0p, v0p + 16), c0);
        o1 = wmma_bf16(pa, load_frag2(v1p, v1p + 16), c1);
    }

    // ---- epilogue: 1/l is already in the right per-lane row order -------
    const int b = bh >> 2;                // /H
    const int h = bh & 3;
#pragma unroll
    for (int v = 0; v < 8; ++v) {
        const int srow   = qt * 16 + v + half * 8;
        const float linv = 1.0f / lrun[v];
        float* orow = out + ((size_t)(b * Ssz + srow)) * Dsz + h * HD;
        orow[l16]      = o0[v] * linv;
        orow[16 + l16] = o1[v] * linv;
    }
}

// ---------------------------------------------------------------------------
extern "C" void kernel_launch(void* const* d_in, const int* in_sizes, int n_in,
                              void* d_out, int out_size, void* d_ws, size_t ws_size,
                              hipStream_t stream) {
    const float* x  = (const float*)d_in[0];
    const float* Wq = (const float*)d_in[1];
    const float* bq = (const float*)d_in[2];
    const float* Wk = (const float*)d_in[3];
    const float* bk = (const float*)d_in[4];
    const float* Wv = (const float*)d_in[5];
    const float* bv = (const float*)d_in[6];

    __bf16* qws  = (__bf16*)d_ws;
    __bf16* kws  = (__bf16*)((char*)d_ws + (size_t)NE * 2);
    __bf16* vtws = (__bf16*)((char*)d_ws + (size_t)NE * 4);

    // 3 matrices * (B*S/16) row tiles = 3072 waves -> 384 blocks of 8 waves.
    qkv_proj_kernel<<<384, 256, 0, stream>>>(x, Wq, bq, Wk, bk, Wv, bv,
                                             qws, kws, vtws);
    // B*H*(S/16) = 4096 waves -> 512 blocks of 8 waves.
    flash_attn_kernel<<<512, 256, 0, stream>>>(qws, kws, vtws, (float*)d_out);
}
